// Gemma4TextMLP_77910706749692
// MI455X (gfx1250) — matrix operand order repaired and checked by reading, compiled, not run on hardware
//
#include <hip/hip_runtime.h>

typedef __bf16 bf16;
typedef __bf16 v16bf __attribute__((ext_vector_type(16)));
typedef __bf16 v8bf  __attribute__((ext_vector_type(8)));
typedef float  v8f   __attribute__((ext_vector_type(8)));
typedef int    v4i_  __attribute__((ext_vector_type(4)));

// Async global->LDS DMA path (gfx1250): guarded so an absent builtin degrades
// to the proven synchronous copy instead of failing the compile.
#if defined(__HIP_DEVICE_COMPILE__) && __has_builtin(__builtin_amdgcn_global_load_async_to_lds_b128)
#define USE_ASYNC 1
#else
#define USE_ASYNC 0
#endif

__device__ __forceinline__ void wait_async_copies() {
#if USE_ASYNC
#if __has_builtin(__builtin_amdgcn_s_wait_asynccnt)
    __builtin_amdgcn_s_wait_asynccnt(0);
#else
    asm volatile("s_wait_asynccnt 0" ::: "memory");
#endif
#endif
}

// copy 16B from global to LDS (async DMA if available, else through VGPRs)
__device__ __forceinline__ void copy16(const bf16* __restrict__ g, bf16* __restrict__ l) {
#if USE_ASYNC
    // builtin signature (probe-confirmed): (v4i AS1*, v4i AS3*, imm offset, imm cpol)
    __builtin_amdgcn_global_load_async_to_lds_b128(
        (__attribute__((address_space(1))) v4i_*)(uintptr_t)g,
        (__attribute__((address_space(3))) v4i_*)l, 0, 0);
#else
    *(uint4*)l = *(const uint4*)g;
#endif
}

// ---------------------------------------------------------------- utilities

__device__ __forceinline__ float gelu_tanh_f(float x) {
    float x3 = x * x * x;
    return 0.5f * x * (1.0f + tanhf(0.79788456080286535588f * (x + 0.044715f * x3)));
}

__global__ void init_amax_kernel(float* amax) {
    if (threadIdx.x < 4) amax[threadIdx.x] = 0.0f;
}

__global__ void amax_kernel(const float* __restrict__ w, size_t n4, float* __restrict__ slot) {
    float m = 0.0f;
    size_t stride = (size_t)gridDim.x * blockDim.x;
    for (size_t i = (size_t)blockIdx.x * blockDim.x + threadIdx.x; i < n4; i += stride) {
        float4 v = ((const float4*)w)[i];
        m = fmaxf(m, fmaxf(fmaxf(fabsf(v.x), fabsf(v.y)), fmaxf(fabsf(v.z), fabsf(v.w))));
    }
    atomicMax((int*)slot, __float_as_int(m));  // non-negative: int cmp == float cmp
}

__global__ void quant8_to_bf16_kernel(const float* __restrict__ w,
                                      const float* __restrict__ amax,
                                      bf16* __restrict__ o, size_t n) {
    size_t i = ((size_t)blockIdx.x * blockDim.x + threadIdx.x) * 4;
    if (i >= n) return;
    float a     = fmaxf(amax[0], 1e-8f);
    float scale = a * (1.0f / 127.0f);
    float inv   = 127.0f / a;
    float4 v = *(const float4*)(w + i);
    float r[4] = {v.x, v.y, v.z, v.w};
#pragma unroll
    for (int j = 0; j < 4; ++j) {
        float q = rintf(r[j] * inv);
        q = fminf(fmaxf(q, -128.0f), 127.0f);
        o[i + j] = (bf16)(q * scale);
    }
}

__global__ void cvt_bf16_kernel(const float* __restrict__ x, bf16* __restrict__ o, size_t n) {
    size_t i = ((size_t)blockIdx.x * blockDim.x + threadIdx.x) * 4;
    if (i >= n) return;
    float4 v = *(const float4*)(x + i);
    o[i + 0] = (bf16)v.x; o[i + 1] = (bf16)v.y;
    o[i + 2] = (bf16)v.z; o[i + 3] = (bf16)v.w;
}

__global__ void hidden_kernel(bf16* __restrict__ gg, const bf16* __restrict__ up,
                              const float* __restrict__ amax_slot, size_t n) {
    float amax  = fmaxf(amax_slot[0], 1e-8f);
    float scale = amax * (1.0f / 32767.0f);
    float inv   = 32767.0f / amax;
    size_t stride = (size_t)gridDim.x * blockDim.x;
    for (size_t i = (size_t)blockIdx.x * blockDim.x + threadIdx.x; i < n; i += stride) {
        float g = (float)gg[i];
        float q = rintf(g * inv);
        q = fminf(fmaxf(q, -32768.0f), 32767.0f);
        gg[i] = (bf16)((q * scale) * (float)up[i]);
    }
}

// ---------------------------------------------------------------- WMMA GEMM
// C[M,N] = A[M,K] * B[N,K]^T, bf16 in, f32 accum.
// 256 threads = 8 waves. Block tile 256x128, BK=32, LDS double-buffered.
// Wave grid 4(m) x 2(n): each wave owns 64x64 = 4x4 WMMA tiles ->
// 16 v_wmma_f32_16x16x32_bf16 per stage against 16 ds_load_b128.
// Async prefetch of stage k+1 overlaps the WMMAs of stage k; one barrier/stage.
template <int MODE>
__global__ __launch_bounds__(256)
void gemm_bf16_wmma_kernel(const bf16* __restrict__ A, const bf16* __restrict__ B,
                           int M, int N, int K,
                           bf16* __restrict__ outb, float* __restrict__ outf,
                           float* __restrict__ amax_slot) {
    __shared__ __attribute__((aligned(32))) bf16 As[2][256 * 32];
    __shared__ __attribute__((aligned(32))) bf16 Bs[2][128 * 32];

    const int tid  = threadIdx.x;
    const int lane = tid & 31;
    const int wave = tid >> 5;
    const int wm   = (wave >> 1) * 64;   // wave row offset in block tile
    const int wn   = (wave & 1) * 64;    // wave col offset in block tile
    const int r    = lane & 15;
    const int hi   = lane >> 4;          // 0 | 1

    const int m0 = blockIdx.y * 256;
    const int n0 = blockIdx.x * 128;

    v8f acc[4][4];
#pragma unroll
    for (int i = 0; i < 4; ++i)
#pragma unroll
        for (int j = 0; j < 4; ++j)
#pragma unroll
            for (int e = 0; e < 8; ++e) acc[i][j][e] = 0.0f;

    // cooperative copy slices:
    //   A tile 256 rows x 32 K: thread t copies row t (64B = 4 x 16B)
    //   B tile 128 rows x 32 K: thread t copies half-row (32B = 2 x 16B)
    const bf16* gA = A + (size_t)(m0 + tid) * K;
    const bf16* gB = B + (size_t)(n0 + (tid >> 1)) * K + (tid & 1) * 16;
    const int   lAo = tid * 32;
    const int   lBo = (tid >> 1) * 32 + (tid & 1) * 16;

    // prologue: stage 0 into buffer 0
#pragma unroll
    for (int c = 0; c < 4; ++c) copy16(gA + c * 8, &As[0][lAo + c * 8]);
#pragma unroll
    for (int c = 0; c < 2; ++c) copy16(gB + c * 8, &Bs[0][lBo + c * 8]);
    wait_async_copies();
    __syncthreads();

    int p = 0;
    for (int k0 = 0; k0 < K; k0 += 32, p ^= 1) {
        // prefetch next stage into the other buffer (safe: that buffer was
        // last read in stage k0-32, fully consumed before the last barrier)
        int kn = k0 + 32;
        if (kn < K) {
#pragma unroll
            for (int c = 0; c < 4; ++c) copy16(gA + kn + c * 8, &As[p ^ 1][lAo + c * 8]);
#pragma unroll
            for (int c = 0; c < 2; ++c) copy16(gB + kn + c * 8, &Bs[p ^ 1][lBo + c * 8]);
        }

        // A fragment (16x32 bf16): lanes 0-15 hold K 0-15, lanes 16-31 K 16-31
        v16bf af[4];
#pragma unroll
        for (int ms = 0; ms < 4; ++ms) {
            const bf16* q = &As[p][(wm + ms * 16 + r) * 32 + hi * 16];
            v8bf lo  = *(const v8bf*)(q);
            v8bf hi8 = *(const v8bf*)(q + 8);
            af[ms] = __builtin_shufflevector(lo, hi8, 0, 1, 2, 3, 4, 5, 6, 7,
                                             8, 9, 10, 11, 12, 13, 14, 15);
        }
        // B fragment (32x16): lane = column, lanes 0-15 K 0-15, lanes 16-31 K 16-31
        v16bf bfr[4];
#pragma unroll
        for (int ns = 0; ns < 4; ++ns)
            bfr[ns] = *(const v16bf*)(&Bs[p][(wn + ns * 16 + r) * 32 + hi * 16]);

#pragma unroll
        for (int ms = 0; ms < 4; ++ms)
#pragma unroll
            for (int ns = 0; ns < 4; ++ns)
                acc[ms][ns] = __builtin_amdgcn_wmma_f32_16x16x32_bf16(
                    false, af[ms], false, bfr[ns], (short)0, acc[ms][ns], false, false);

        wait_async_copies();   // prefetch landed
        __syncthreads();       // all waves done with buffer p; buffer p^1 visible
    }

    // epilogue: C layout -> lane n = lane%16, VGPR v -> m = v + 8*(lane>>4)
    float tmax = 0.0f;
#pragma unroll
    for (int ms = 0; ms < 4; ++ms) {
#pragma unroll
        for (int ns = 0; ns < 4; ++ns) {
#pragma unroll
            for (int v = 0; v < 8; ++v) {
                int m = m0 + wm + ms * 16 + hi * 8 + v;
                int n = n0 + wn + ns * 16 + r;
                size_t idx = (size_t)m * N + n;
                float val = acc[ms][ns][v];
                if (MODE == 0) {
                    float g = gelu_tanh_f(val);
                    tmax = fmaxf(tmax, fabsf(g));
                    outb[idx] = (bf16)g;
                } else if (MODE == 1) {
                    outb[idx] = (bf16)val;
                } else {
                    outf[idx] = val;
                }
            }
        }
    }
    if (MODE == 0) atomicMax((int*)amax_slot, __float_as_int(tmax));
}

// ---------------------------------------------------------------- launcher

extern "C" void kernel_launch(void* const* d_in, const int* in_sizes, int n_in,
                              void* d_out, int out_size, void* d_ws, size_t ws_size,
                              hipStream_t stream) {
    const float* x  = (const float*)d_in[0];
    const float* wg = (const float*)d_in[1];
    const float* wu = (const float*)d_in[2];
    const float* wd = (const float*)d_in[3];
    float* out = (float*)d_out;

    const int H = 2048;
    const int M = in_sizes[0] / H;   // 8192 tokens
    const int I = in_sizes[1] / H;   // 8192
    const size_t nx = (size_t)in_sizes[0];
    const size_t ng = (size_t)in_sizes[1];
    const size_t nu = (size_t)in_sizes[2];
    const size_t nd = (size_t)in_sizes[3];
    const size_t nh = (size_t)M * I;

    // workspace layout
    float* amax = (float*)d_ws;          // [0]=wg [1]=wu [2]=wd [3]=amax(gelu(gate))
    bf16* xb  = (bf16*)((char*)d_ws + 256);
    bf16* wgb = xb  + nx;
    bf16* wub = wgb + ng;
    bf16* wdb = wub + nu;
    bf16* gg  = wdb + nd;                // gelu(gate) -> hidden (in place)
    bf16* upb = gg  + nh;

    init_amax_kernel<<<1, 32, 0, stream>>>(amax);
    amax_kernel<<<512, 256, 0, stream>>>(wg, ng / 4, amax + 0);
    amax_kernel<<<512, 256, 0, stream>>>(wu, nu / 4, amax + 1);
    amax_kernel<<<512, 256, 0, stream>>>(wd, nd / 4, amax + 2);

    quant8_to_bf16_kernel<<<(int)(ng / 1024), 256, 0, stream>>>(wg, amax + 0, wgb, ng);
    quant8_to_bf16_kernel<<<(int)(nu / 1024), 256, 0, stream>>>(wu, amax + 1, wub, nu);
    quant8_to_bf16_kernel<<<(int)(nd / 1024), 256, 0, stream>>>(wd, amax + 2, wdb, nd);
    cvt_bf16_kernel<<<(int)(nx / 1024), 256, 0, stream>>>(x, xb, nx);

    dim3 blk(256);
    // gate = gelu(x Wg^T) -> bf16 + global amax
    gemm_bf16_wmma_kernel<0><<<dim3(I / 128, M / 256), blk, 0, stream>>>(
        xb, wgb, M, I, H, gg, nullptr, amax + 3);
    // up = x Wu^T -> bf16
    gemm_bf16_wmma_kernel<1><<<dim3(I / 128, M / 256), blk, 0, stream>>>(
        xb, wub, M, I, H, upb, nullptr, nullptr);
    // hidden = fq16(gelu_gate) * up   (in place into gg)
    hidden_kernel<<<4096, 256, 0, stream>>>(gg, upb, amax + 3, nh);
    // out = hidden Wd^T -> f32
    gemm_bf16_wmma_kernel<2><<<dim3(H / 128, M / 256), blk, 0, stream>>>(
        gg, wdb, M, H, I, nullptr, out, nullptr);
}
